// Layer1D_8022998909160
// MI455X (gfx1250) — matrix operand order repaired and checked by reading, not compiled, not run on hardware
//
#include <hip/hip_runtime.h>
#include <stdint.h>

// Problem constants (from reference setup_inputs)
#define CIN    256
#define COUT   512
#define LEN    1024
#define TSTEPS 4
#define BATCH  16
#define KW     3

// Tiling
#define MT 128                 // Cout per workgroup
#define NT 64                  // L per workgroup
#define XCOLS (NT + 2)         // halo for K=3, pad=1
#define XPITCH (CIN + 8)       // LDS pitch in bf16 elems: 512B data + 16B pad = 528B
#define XROWS (LEN + 2)        // transposed x rows per (b,t): 1 zero halo row each side

typedef __attribute__((ext_vector_type(16))) __bf16 v16bf;
typedef __attribute__((ext_vector_type(8)))  __bf16 v8bf;
typedef __attribute__((ext_vector_type(8)))  float  v8f;
typedef __attribute__((ext_vector_type(4)))  unsigned int u32x4;
typedef __attribute__((ext_vector_type(8)))  int i32x8;
typedef __attribute__((ext_vector_type(4)))  int i32x4;

// ---- CDNA5 data-mover feature detection ------------------------------------
#if defined(__has_builtin)
# if __has_builtin(__builtin_amdgcn_tensor_load_to_lds)
#  define USE_TDM 1
# endif
# if !defined(USE_TDM) && __has_builtin(__builtin_amdgcn_global_load_async_to_lds_b128)
#  define USE_ASYNC 1
# endif
#endif
#if defined(USE_TDM) && __has_include(<hip/amd_detail/amd_gfx1250_TDM.h>)
# define TDM_6ARG 1
#endif

// ---------------- prep kernels ----------------

// Transpose + convert: x [bt][cin][l] f32  ->  xbT [bt][l+1][cin] bf16
// (row 0 and row 1025 of each (b,t) slab are zero halo rows)
__global__ __launch_bounds__(256) void cvt_x_bf16_T(const float* __restrict__ x,
                                                    __bf16* __restrict__ xbT) {
  size_t i = (size_t)blockIdx.x * 256 + threadIdx.x;   // 16,777,216 total
  int l   = (int)(i & (LEN - 1));
  int cin = (int)((i >> 10) & (CIN - 1));
  int bt  = (int)(i >> 18);
  xbT[((size_t)bt * XROWS + l + 1) * CIN + cin] = (__bf16)x[i];
}

__global__ __launch_bounds__(256) void zero_halo_rows(__bf16* __restrict__ xbT) {
  int i   = blockIdx.x * 256 + threadIdx.x;            // 64*2*256 = 32768 total
  int cin = i & (CIN - 1);
  int r   = (i >> 8) & 1;
  int bt  = i >> 9;
  size_t row = (size_t)bt * XROWS + (r ? (XROWS - 1) : 0);
  xbT[row * CIN + cin] = (__bf16)0.f;
}

// Re-swizzle conv_w [Cout][Cin][K] (f32) into A-fragment-native bf16 layout:
// frag id f = (tap*32 + mtile)*8 + cinblock ; storage: f*512 + lane*16 + e
__global__ __launch_bounds__(256) void prep_wfrag(const float* __restrict__ w,
                                                  __bf16* __restrict__ wf) {
  int idx  = blockIdx.x * 256 + threadIdx.x;           // total 393216
  int e    = idx & 15;
  int lane = (idx >> 4) & 31;
  int f    = idx >> 9;
  int cb   = f & 7;
  int mt   = (f >> 3) & 31;
  int tap  = f >> 8;
  int m    = mt * 16 + (lane & 15);
  int kk   = 16 * (lane >> 4) + e;
  int cin  = cb * 32 + kk;
  wf[idx]  = (__bf16)w[(m * CIN + cin) * KW + tap];
}

// Fold conv bias + BN: y = conv*inv + ((conv_b - mean)*inv + beta)
__global__ __launch_bounds__(256) void prep_scale_bias(const float* __restrict__ cb,
                                                       const float* __restrict__ g,
                                                       const float* __restrict__ be,
                                                       const float* __restrict__ mu,
                                                       const float* __restrict__ vr,
                                                       float* __restrict__ scale,
                                                       float* __restrict__ bias) {
  int c = blockIdx.x * 256 + threadIdx.x;
  if (c < COUT) {
    float inv = g[c] * rsqrtf(vr[c] + 1e-5f);
    scale[c] = inv;
    bias[c]  = (cb[c] - mu[c]) * inv + be[c];
  }
}

// ---------------- main fused conv+BN+LIF kernel ----------------

__device__ inline v16bf load_frag16(const __bf16* p) {
  v8bf lo = *(const v8bf*)p;
  v8bf hi = *(const v8bf*)(p + 8);
  return __builtin_shufflevector(lo, hi, 0, 1, 2, 3, 4, 5, 6, 7,
                                 8, 9, 10, 11, 12, 13, 14, 15);
}

__global__ __launch_bounds__(256)
void snn_conv_lif(const __bf16* __restrict__ xbT,
                  const __bf16* __restrict__ wfrag,
                  const float* __restrict__ scale,
                  const float* __restrict__ bias,
                  float* __restrict__ out) {
  __shared__ __bf16 xs[XCOLS * XPITCH];   // input tile: [col][cin], 528B pitch
  __shared__ float  ss[MT];
  __shared__ float  sb[MT];

  const int tid   = threadIdx.x;
  const int lane  = tid & 31;
  const int wave  = tid >> 5;        // 0..7
  const int wm    = wave >> 1;       // 0..3 : 32-row group within 128
  const int wn    = wave & 1;        // 0..1 : 32-col group within 64
  const int ln15  = lane & 15;
  const int lhi   = lane >> 4;       // 0/1

  const int b     = blockIdx.z;
  const int cout0 = blockIdx.y * MT;
  const int l0    = blockIdx.x * NT;

  if (tid < MT) {
    ss[tid] = scale[cout0 + tid];
    sb[tid] = bias[cout0 + tid];
  }

  v8f mem[2][2];
  v8f zero = {0.f, 0.f, 0.f, 0.f, 0.f, 0.f, 0.f, 0.f};
#pragma unroll
  for (int tm = 0; tm < 2; ++tm)
#pragma unroll
    for (int tn = 0; tn < 2; ++tn) mem[tm][tn] = zero;

  for (int t = 0; t < TSTEPS; ++t) {
    __syncthreads();   // previous iteration done reading xs (t=0: ss/sb ready)

    // ---- stage x tile into LDS: rows are contiguous 512B in xbT; tile row 0
    //      corresponds to l0-1 (== transposed row l0 thanks to the halo shift)
    const size_t tile_row0 = (size_t)(b * TSTEPS + t) * XROWS + l0;
#if defined(USE_TDM)
    if (wave == 0) {
      // ---- Tensor Data Mover: one 2-D tile DMA with LDS pitch padding -----
      unsigned long long gaddr =
          (unsigned long long)(uintptr_t)xbT + tile_row0 * (CIN * 2);
      unsigned lds_base = (unsigned)(uintptr_t)(&xs[0]);
      u32x4 g0;
      g0[0] = 1u;                                        // count=1, user mode
      g0[1] = lds_base;                                  // lds_addr
      g0[2] = (unsigned)gaddr;                           // global_addr[31:0]
      g0[3] = (unsigned)((gaddr >> 32) & 0x01FFFFFFu) | (2u << 30); // addr[56:32], type=2
      i32x8 g1;
      g1[0] = (int)((1u << 20) | (6u << 22) | (3u << 25)); // data_size=1B, pad_en,
                                                           // pad every 128 DW, pad 4 DW
      g1[1] = (int)(512u << 16);       // tensor_dim0[15:0] = 512 (bytes/row)
      g1[2] = (int)(1026u << 16);      // tensor_dim0 hi=0 | tensor_dim1 lo = 1026
      g1[3] = (int)(512u << 16);       // tensor_dim1 hi=0 | tile_dim0 = 512
      g1[4] = (int)XCOLS;              // tile_dim1 = 66, tile_dim2 = 0
      g1[5] = (int)512;                // tensor_dim0_stride lo32 = 512
      g1[6] = (int)(512u << 16);       // td0_stride hi=0 | tensor_dim1_stride lo = 512
      g1[7] = 0;                       // tensor_dim1_stride hi = 0
      i32x4 z4 = {0, 0, 0, 0};
#if defined(TDM_6ARG)
      i32x8 z8 = {0, 0, 0, 0, 0, 0, 0, 0};
      __builtin_amdgcn_tensor_load_to_lds(g0, g1, z4, z4, z8, 0);
#else
      __builtin_amdgcn_tensor_load_to_lds(g0, g1, z4, z4, 0);
#endif
      __builtin_amdgcn_s_wait_tensorcnt(0);
    }
#elif defined(USE_ASYNC)
    for (int i = tid; i < XCOLS * (CIN / 8); i += 256) {
      int col = i >> 5;                 // CIN/8 == 32 segments of 16B
      int seg = i & 31;
      const __bf16* src = xbT + (tile_row0 + col) * CIN + seg * 8;
      __bf16* dst = &xs[col * XPITCH + seg * 8];
      __builtin_amdgcn_global_load_async_to_lds_b128((void*)src, (void*)dst, 0, 0);
    }
    __builtin_amdgcn_s_wait_asynccnt(0);
#else
    for (int i = tid; i < XCOLS * (CIN / 8); i += 256) {
      int col = i >> 5;
      int seg = i & 31;
      const v8bf* src = (const v8bf*)(xbT + (tile_row0 + col) * CIN + seg * 8);
      *(v8bf*)&xs[col * XPITCH + seg * 8] = *src;
    }
#endif
    __syncthreads();

    // ---- GEMM: D[m][n] += sum_{tap,cin} W[m][cin][tap] * x[cin][n + tap - 1]
    v8f acc[2][2];
#pragma unroll
    for (int tm = 0; tm < 2; ++tm)
#pragma unroll
      for (int tn = 0; tn < 2; ++tn) acc[tm][tn] = zero;

    for (int tap = 0; tap < KW; ++tap) {
#pragma unroll
      for (int cb = 0; cb < CIN / 32; ++cb) {
        v16bf a[2];
#pragma unroll
        for (int tm = 0; tm < 2; ++tm) {
          int mt = (cout0 >> 4) + wm * 2 + tm;
          const __bf16* ap =
              wfrag + (((size_t)((tap * (COUT / 16) + mt) * (CIN / 32) + cb)) << 9) +
              lane * 16;
          a[tm] = load_frag16(ap);                 // coalesced 32B per lane
        }
        v16bf bmat[2];
#pragma unroll
        for (int tn = 0; tn < 2; ++tn) {
          int col = wn * 32 + tn * 16 + ln15 + tap;
          const __bf16* bp = &xs[col * XPITCH + cb * 32 + lhi * 16];
          bmat[tn] = load_frag16(bp);              // two ds_load_b128 per lane
        }
#pragma unroll
        for (int tm = 0; tm < 2; ++tm)
#pragma unroll
          for (int tn = 0; tn < 2; ++tn)
            acc[tm][tn] = __builtin_amdgcn_wmma_f32_16x16x32_bf16(
                false, a[tm], false, bmat[tn], (short)0, acc[tm][tn], false, false);
      }
    }

    // ---- epilogue: BN scale/bias, LIF update (membrane in VGPRs), store spikes
    float* outbt = out + (size_t)(b * TSTEPS + t) * COUT * LEN;
#pragma unroll
    for (int tm = 0; tm < 2; ++tm) {
#pragma unroll
      for (int tn = 0; tn < 2; ++tn) {
#pragma unroll
        for (int r = 0; r < 8; ++r) {
          int mloc = wm * 32 + tm * 16 + r + 8 * lhi;    // 0..127
          int m    = cout0 + mloc;
          int n    = l0 + wn * 32 + tn * 16 + ln15;
          float y  = acc[tm][tn][r] * ss[mloc] + sb[mloc];
          float mv = mem[tm][tn][r] * 0.5f + y;
          float sp = (mv - 1.0f > 0.0f) ? 1.0f : 0.0f;
          mem[tm][tn][r] = (1.0f - sp) * mv;
          outbt[(size_t)m * LEN + n] = sp;
        }
      }
    }
  }
}

// ---------------- launch ----------------

extern "C" void kernel_launch(void* const* d_in, const int* in_sizes, int n_in,
                              void* d_out, int out_size, void* d_ws, size_t ws_size,
                              hipStream_t stream) {
  const float* x      = (const float*)d_in[0];
  const float* conv_w = (const float*)d_in[1];
  const float* conv_b = (const float*)d_in[2];
  const float* gamma  = (const float*)d_in[3];
  const float* beta   = (const float*)d_in[4];
  const float* mean   = (const float*)d_in[5];
  const float* var    = (const float*)d_in[6];
  float* out = (float*)d_out;

  const size_t n_x  = (size_t)BATCH * TSTEPS * CIN * LEN;          // 16,777,216
  const size_t n_xT = (size_t)BATCH * TSTEPS * XROWS * CIN;        // 16,809,984
  const size_t n_w  = (size_t)KW * COUT * CIN;                     // 393,216

  char* ws = (char*)d_ws;
  __bf16* xbT   = (__bf16*)ws;                                     // ~32 MB
  __bf16* wfrag = (__bf16*)(ws + n_xT * sizeof(__bf16));           // 768 KB
  float*  scale = (float*)(ws + n_xT * sizeof(__bf16) + n_w * sizeof(__bf16));
  float*  bias  = scale + COUT;

  cvt_x_bf16_T<<<(int)(n_x / 256), 256, 0, stream>>>(x, xbT);
  zero_halo_rows<<<(BATCH * TSTEPS * 2 * CIN) / 256, 256, 0, stream>>>(xbT);
  prep_wfrag<<<(int)(n_w / 256), 256, 0, stream>>>(conv_w, wfrag);
  prep_scale_bias<<<2, 256, 0, stream>>>(conv_b, gamma, beta, mean, var, scale, bias);

  dim3 grid(LEN / NT, COUT / MT, BATCH);   // (16, 4, 16)
  snn_conv_lif<<<grid, 256, 0, stream>>>(xbT, wfrag, scale, bias, out);
}
